// BahdanauAttention_25228637896981
// MI455X (gfx1250) — matrix operand order repaired and checked by reading, not compiled, not run on hardware
//
#include <hip/hip_runtime.h>
#include <hip/hip_bf16.h>

// ---------------------------------------------------------------------------
// Bahdanau attention, CDNA5 (gfx1250), wave32.
//   B=8, T=256, S=256, H=512
//   Phase 0: convert enc/query -> bf16; transpose+convert W_h/W_s -> bf16
//   Phase 1: EH = enc @ W_h^T, QS = q @ W_s^T   (bf16 WMMA, f32 accum, 2x4 tiles)
//   Phase 2: score = sum_h v*tanh(eh+qs); softmax -> attn (bf16)
//   Phase 3: context = attn @ enc                (bf16 WMMA, f32 accum)
// All GEMM dims are compile-time template constants so store addressing folds
// into immediate offsets and index math becomes shifts on constants.
// ---------------------------------------------------------------------------

typedef __attribute__((ext_vector_type(16))) __bf16 v16bf;
typedef __attribute__((ext_vector_type(8)))  __bf16 v8bf;
typedef __attribute__((ext_vector_type(8)))  float  v8f;

#if __has_builtin(__builtin_amdgcn_tanhf)
#define FAST_TANH(x) __builtin_amdgcn_tanhf(x)
#elif __has_builtin(__builtin_amdgcn_tanh_f32)
#define FAST_TANH(x) __builtin_amdgcn_tanh_f32(x)
#else
#define FAST_TANH(x) tanhf(x)
#endif

static constexpr int Bn = 8;
static constexpr int Tn = 256;
static constexpr int Sn = 256;
static constexpr int Hn = 512;

// ---------------------------------------------------------------------------
// bf16 fragment loaders (A via b128 loads, B via strided element loads; no conversion in the hot loop)
// A fragment 16x32 (MxK): lane L -> m = m0 + L%16 ; half = L/16 ;
//   elements 0-7 at k0+half*8, elements 8-15 at k0+half*8+16.
// ---------------------------------------------------------------------------
template <int LD>
__device__ __forceinline__ v16bf load_fragA_bf16(const __bf16* __restrict__ base,
                                                 int m0, int k0, int lane) {
    const __bf16* p = base + (size_t)(m0 + (lane & 15)) * LD + k0 + ((lane >> 4) << 3);
    v8bf lo = *(const v8bf*)p;          // 16B
    v8bf hi = *(const v8bf*)(p + 16);   // 16B
    return __builtin_shufflevector(lo, hi, 0, 1, 2, 3, 4, 5, 6, 7,
                                           8, 9, 10, 11, 12, 13, 14, 15);
}

// B fragment 32x16 (KxN), row-major [K][N]: lane L -> n = n0 + L%16 ; half = L/16 ;
//   elements 0-7 at k0+half*8, elements 8-15 at k0+half*8+16 (strided loads).
template <int LD>
__device__ __forceinline__ v16bf load_fragB_bf16(const __bf16* __restrict__ Bm,
                                                 int n0, int k0, int lane) {
    const __bf16* p = Bm + (size_t)(k0 + ((lane >> 4) << 3)) * LD + n0 + (lane & 15);
    v16bf r;
#pragma unroll
    for (int i = 0; i < 8; ++i) {
        r[i]     = p[(size_t)i * LD];
        r[i + 8] = p[(size_t)(i + 16) * LD];
    }
    return r;
}

__device__ __forceinline__ v8f wmma_bf16(v16bf a, v16bf b, v8f c) {
    return __builtin_amdgcn_wmma_f32_16x16x32_bf16(
        /*neg_a=*/false, a, /*neg_b=*/false, b,
        /*c_mod=*/(short)0, c, /*reuse_a=*/false, /*reuse_b=*/false);
}

// ---------------------------------------------------------------------------
// Phase 0a: f32 -> bf16 copy, 8 elements / thread (b128-in, b128-out)
// ---------------------------------------------------------------------------
__global__ __launch_bounds__(256)
void cvt_bf16_kernel(const float* __restrict__ src, __bf16* __restrict__ dst, int n8) {
    const int i = blockIdx.x * 256 + threadIdx.x;
    if (i >= n8) return;
    const float4* p = (const float4*)src + (size_t)i * 2;
    const float4 a = p[0], b = p[1];
    v8bf o;
    o[0] = (__bf16)a.x; o[1] = (__bf16)a.y; o[2] = (__bf16)a.z; o[3] = (__bf16)a.w;
    o[4] = (__bf16)b.x; o[5] = (__bf16)b.y; o[6] = (__bf16)b.z; o[7] = (__bf16)b.w;
    ((v8bf*)dst)[i] = o;
}

// ---------------------------------------------------------------------------
// Phase 0b: W[N][K] f32 -> Wt[K][N] bf16, 32x32 LDS tiles (both sides coalesced)
// ---------------------------------------------------------------------------
__global__ __launch_bounds__(256)
void transpose_cvt_kernel(const float* __restrict__ W, __bf16* __restrict__ Wt) {
    __shared__ float tile[32][33];
    const int bx = blockIdx.x & 15, by = blockIdx.x >> 4;   // 512/32 = 16 tiles/dim
    const int x0 = bx << 5, y0 = by << 5;
    const int tx = threadIdx.x & 31, tw = threadIdx.x >> 5; // 32 x 8
#pragma unroll
    for (int r = tw; r < 32; r += 8)
        tile[r][tx] = W[(size_t)(y0 + r) * Hn + x0 + tx];
    __syncthreads();
#pragma unroll
    for (int r = tw; r < 32; r += 8)
        Wt[(size_t)(x0 + r) * Hn + y0 + tx] = (__bf16)tile[tx][r];
}

// ---------------------------------------------------------------------------
// Batched bf16 GEMM: C[M][N] = A[M][K] @ B[K][N], f32 accumulate.
// One wave owns a 32x64 C block (2 M-tiles x 4 N-tiles) -> 8 WMMAs / k-step,
// fed by 4 b128 A loads + 64 strided bf16 B loads. 8 waves / block. Grid sized exactly (no tail).
// M,N,K and batch strides are compile-time so the epilogue uses one base
// address + immediate offsets per C row.
// ---------------------------------------------------------------------------
template <int M, int N, int K, long SA, long SB, long SC>
__global__ __launch_bounds__(256)
void gemm_bf16_wmma_kernel(const __bf16* __restrict__ A, const __bf16* __restrict__ Bmat,
                           float* __restrict__ C) {
    const int lane = threadIdx.x & 31;
    const int wave = threadIdx.x >> 5;
    const int w    = blockIdx.x * 8 + wave;
    constexpr int wavesN = N >> 6;              // 64-wide N blocks
    constexpr int wavesM = M >> 5;              // 32-tall M blocks
    constexpr int wavesPerBatch = wavesM * wavesN;
    const int batch = w / wavesPerBatch;
    const int rem   = w % wavesPerBatch;
    const int m0 = (rem / wavesN) << 5;
    const int n0 = (rem % wavesN) << 6;

    const __bf16* Ab = A    + (size_t)batch * SA;
    const __bf16* Bb = Bmat + (size_t)batch * SB;
    float*        Cb = C    + (size_t)batch * SC;

    v8f acc[2][4] = {};
#pragma unroll 1
    for (int k0 = 0; k0 < K; k0 += 32) {
        v16bf a[2], b[4];
#pragma unroll
        for (int i = 0; i < 2; ++i)
            a[i] = load_fragA_bf16<K>(Ab, m0 + i * 16, k0, lane);
#pragma unroll
        for (int j = 0; j < 4; ++j)
            b[j] = load_fragB_bf16<N>(Bb, n0 + j * 16, k0, lane);
#pragma unroll
        for (int i = 0; i < 2; ++i)
#pragma unroll
            for (int j = 0; j < 4; ++j)
                acc[i][j] = wmma_bf16(a[i], b[j], acc[i][j]);
    }

    // C/D layout: lane -> n = lane%16 ; VGPR r -> m = r + (lane/16)*8
    const int nc = lane & 15;
    const int mb = (lane >> 4) << 3;
#pragma unroll
    for (int i = 0; i < 2; ++i) {
        float* dst = Cb + (size_t)(m0 + i * 16 + mb) * N + n0 + nc;
#pragma unroll
        for (int j = 0; j < 4; ++j)
#pragma unroll
            for (int r = 0; r < 8; ++r)
                dst[(size_t)r * N + j * 16] = acc[i][j][r];   // const offsets
    }
}

// ---------------------------------------------------------------------------
// Phase 2: score + softmax. One block per (b, 16-row t tile): grid = 128.
// qs tile (16x512) + v staged in LDS; eh rows streamed (L2 resident).
// Emits attn weights as bf16, ready to be the WMMA A operand of phase 3.
// ---------------------------------------------------------------------------
__global__ __launch_bounds__(256)
void score_softmax_kernel(const float* __restrict__ EH, const float* __restrict__ QS,
                          const float* __restrict__ vvec,
                          const unsigned char* __restrict__ mask,
                          __bf16* __restrict__ ATTNB) {
    __shared__ float qsTile[16 * Hn];   // 32 KB
    __shared__ float sc[16 * Sn];       // 16 KB
    __shared__ float vsh[Hn];           //  2 KB

    const int b  = blockIdx.x >> 4;
    const int t0 = (blockIdx.x & 15) << 4;

    for (int i = threadIdx.x; i < 16 * Hn; i += 256) {
        const int tt = i >> 9, h = i & (Hn - 1);
        qsTile[i] = QS[(size_t)(b * Tn + t0 + tt) * Hn + h];
    }
    for (int i = threadIdx.x; i < Hn; i += 256) vsh[i] = vvec[i];
    __syncthreads();

    const int lane = threadIdx.x & 31;
    const int wave = threadIdx.x >> 5;
    const float* ehB = EH + (size_t)b * Sn * Hn;

    for (int tt = wave; tt < 16; tt += 8) {
        const float* q = qsTile + tt * Hn;
        for (int s = 0; s < Sn; ++s) {
            const float* eh = ehB + (size_t)s * Hn;
            if (s + 1 < Sn)
                __builtin_prefetch(ehB + (size_t)(s + 1) * Hn + lane * 16, 0, 0);
            float sum = 0.f;
#pragma unroll 4
            for (int h = lane; h < Hn; h += 32)
                sum = fmaf(vsh[h], FAST_TANH(eh[h] + q[h]), sum);
#pragma unroll
            for (int off = 16; off; off >>= 1)
                sum += __shfl_down(sum, off);
            if (lane == 0)
                sc[tt * Sn + s] = mask[b * Sn + s] ? sum : -__builtin_inff();
        }
    }
    __syncthreads();

    for (int tt = wave; tt < 16; tt += 8) {
        float* row = sc + tt * Sn;
        float m = -__builtin_inff();
        for (int j = lane; j < Sn; j += 32) m = fmaxf(m, row[j]);
#pragma unroll
        for (int off = 16; off; off >>= 1) m = fmaxf(m, __shfl_xor(m, off));
        float sum = 0.f;
        for (int j = lane; j < Sn; j += 32) {
            const float e = __expf(row[j] - m);
            row[j] = e;
            sum += e;
        }
#pragma unroll
        for (int off = 16; off; off >>= 1) sum += __shfl_xor(sum, off);
        const float inv = 1.f / sum;
        __bf16* dst = ATTNB + (size_t)(b * Tn + t0 + tt) * Sn;
        for (int j = lane; j < Sn; j += 32) dst[j] = (__bf16)(row[j] * inv);
    }
}

// ---------------------------------------------------------------------------

extern "C" void kernel_launch(void* const* d_in, const int* in_sizes, int n_in,
                              void* d_out, int out_size, void* d_ws, size_t ws_size,
                              hipStream_t stream) {
    const float*         enc  = (const float*)d_in[0];          // [B,S,H]
    const float*         qry  = (const float*)d_in[1];          // [B,T,H]
    const unsigned char* mask = (const unsigned char*)d_in[2];  // [B,S] bool
    const float*         W_h  = (const float*)d_in[3];          // [H,H]
    const float*         W_s  = (const float*)d_in[4];          // [H,H]
    const float*         v    = (const float*)d_in[5];          // [H]
    float* out = (float*)d_out;                                 // [B,T,H]

    // workspace layout (256B-aligned chunks)
    char* ws = (char*)d_ws;
    float*  EH    = (float*)ws;                       ws += (size_t)Bn * Sn * Hn * 4;   // 4 MB
    float*  QS    = (float*)ws;                       ws += (size_t)Bn * Tn * Hn * 4;   // 4 MB
    __bf16* ENCB  = (__bf16*)ws;                      ws += (size_t)Bn * Sn * Hn * 2;   // 2 MB
    __bf16* QB    = (__bf16*)ws;                      ws += (size_t)Bn * Tn * Hn * 2;   // 2 MB
    __bf16* WhT   = (__bf16*)ws;                      ws += (size_t)Hn * Hn * 2;        // 512 KB
    __bf16* WsT   = (__bf16*)ws;                      ws += (size_t)Hn * Hn * 2;        // 512 KB
    __bf16* ATTNB = (__bf16*)ws;                      // B*T*S bf16, 1 MB

    constexpr int M  = Bn * Sn;             // 2048 rows for both projections
    constexpr int n8 = (M * Hn) / 8;        // 131072 vec8 groups

    // Phase 0: precision/layout prep
    cvt_bf16_kernel<<<n8 / 256, 256, 0, stream>>>(enc, ENCB, n8);
    cvt_bf16_kernel<<<n8 / 256, 256, 0, stream>>>(qry, QB, n8);
    transpose_cvt_kernel<<<256, 256, 0, stream>>>(W_h, WhT);
    transpose_cvt_kernel<<<256, 256, 0, stream>>>(W_s, WsT);

    // Phase 1: projections. (2048/32)*(512/64) = 512 waves -> 64 blocks
    gemm_bf16_wmma_kernel<M, Hn, Hn, 0, 0, 0>
        <<<64, 256, 0, stream>>>(ENCB, WhT, EH);
    gemm_bf16_wmma_kernel<M, Hn, Hn, 0, 0, 0>
        <<<64, 256, 0, stream>>>(QB, WsT, QS);

    // Phase 2: additive score (v_tanh_f32) + softmax -> bf16 attn
    score_softmax_kernel<<<Bn * (Tn / 16), 256, 0, stream>>>(EH, QS, v, mask, ATTNB);

    // Phase 3: context = attn @ enc, batched over B.
    // per-b (256/32)*(512/64) = 64 waves; 8 batches -> 512 waves -> 64 blocks
    gemm_bf16_wmma_kernel<Tn, Hn, Sn, (long)Tn * Sn, (long)Sn * Hn, (long)Tn * Hn>
        <<<64, 256, 0, stream>>>(ATTNB, ENCB, out);
}
